// GRUScratch_9208409883433
// MI455X (gfx1250) — matrix operand order repaired and checked by reading, not compiled, not run on hardware
//
#include <hip/hip_runtime.h>
#include <hip/hip_bf16.h>

// ---------------------------------------------------------------------------
// GRU on MI455X (gfx1250), wave32 + v_wmma_f32_16x16x32_bf16 (f32 accumulate).
// Fused formulation: gate = act([x_t ; H] @ [[W_x];[W_h]] + b), K = 2048.
//
// All GEMM-A operands (x, H, R*H) and all weights are kept in the CDNA5
// bf16 K-pair-packed layout so fragment loads are contiguous b128 loads with
// no in-loop conversion:
//   - weights: packed column-major Wt[n][kp] once per call (L2-resident, 12MiB)
//   - x:       packed once per call (64 MiB in ws)
//   - H, R*H:  packed at production time via __shfl_xor with the neighbor lane
// Persistent kernel, 64 blocks x 128 threads, 2 grid barriers per step.
// ---------------------------------------------------------------------------

typedef __attribute__((ext_vector_type(16))) __bf16 v16bf;
typedef __attribute__((ext_vector_type(8)))  float  v8f;
typedef unsigned int u32;
typedef __attribute__((ext_vector_type(8))) u32 u32x8;
typedef __attribute__((ext_vector_type(4))) u32 u32x4;
typedef __attribute__((ext_vector_type(2))) __bf16 bf16x2;

#define TSTEPS 512
#define BATCH  64
#define HID    1024            // hidden size == input size
#define KPH    (HID / 2)       // K-pairs per 1024 segment (512)
#define NBLOCKS 64             // grid blocks for recurrent kernel
#define GRUTHREADS 128         // 4 waves per block

static __device__ inline u32 pack2(float lo, float hi) {
  bf16x2 p;
  p.x = (__bf16)lo;
  p.y = (__bf16)hi;
  return __builtin_bit_cast(u32, p);
}

// A fragment (16x32 bf16, MxK) from a K-pair-packed row (row stride = 512 dw).
// ISA 7.12.2 16-bit A layout: VGPR j holds K-pair kp = kc/2 + j + (j>=4?4:0)
// + 4*half  ->  dwords [kp0 .. kp0+3] and [kp0+8 .. kp0+11], two b128 loads.
static __device__ inline v16bf load_a_packed(const u32* rowptr, int kc, int half) {
  const int kp0 = (kc >> 1) + 4 * half;
  u32x4 lo = *(const u32x4*)(rowptr + kp0);
  u32x4 hi = *(const u32x4*)(rowptr + kp0 + 8);
  u32x8 d = __builtin_shufflevector(lo, hi, 0, 1, 2, 3, 4, 5, 6, 7);
  return __builtin_bit_cast(v16bf, d);
}

// B fragment (32x16 bf16, KxN) from column-major packed weights Wt[n][kp]:
// VGPR j holds K-pair kp0 + j + (j>=4?4:0) with kp0 = kc/2 + 4*half -> two b128 loads.
static __device__ inline v16bf load_b_packed(const u32* colptr, int kc, int half) {
  const int kp0 = (kc >> 1) + 4 * half;
  u32x4 lo = *(const u32x4*)(colptr + kp0);
  u32x4 hi = *(const u32x4*)(colptr + kp0 + 8);
  u32x8 d = __builtin_shufflevector(lo, hi, 0, 1, 2, 3, 4, 5, 6, 7);
  return __builtin_bit_cast(v16bf, d);
}

static __device__ inline v8f wmma_bf16(v16bf a, v16bf b, v8f c) {
  return __builtin_amdgcn_wmma_f32_16x16x32_bf16(false, a, false, b,
                                                 (short)0, c, false, false);
}

static __device__ inline void grid_barrier(u32* bar, int idx) {
  __syncthreads();
  __threadfence();
  if (threadIdx.x == 0) {
    __hip_atomic_fetch_add(bar + idx, 1u, __ATOMIC_ACQ_REL, __HIP_MEMORY_SCOPE_AGENT);
    while (__hip_atomic_load(bar + idx, __ATOMIC_ACQUIRE, __HIP_MEMORY_SCOPE_AGENT)
           < (u32)NBLOCKS) {
      __builtin_amdgcn_s_sleep(1);
    }
  }
  __syncthreads();
}

// ---------------------------------------------------------------------------
// Pack [W_x ; W_h] (2048x1024 f32) -> column-major K-pair bf16: Wt[n][kp],
// kp in [0,1024). LDS 64x64 tile transpose keeps reads AND writes coalesced.
// blockIdx.z selects the gate (0=z, 1=r, 2=h).
// ---------------------------------------------------------------------------
__global__ __launch_bounds__(256) void pack_weights(
    const float* Wxz, const float* Whz,
    const float* Wxr, const float* Whr,
    const float* Wxh, const float* Whh,
    u32* Wzp, u32* Wrp, u32* Whp) {
  __shared__ u32 tile[64][65];
  const int g = blockIdx.z;
  const float* Wx = (g == 0) ? Wxz : ((g == 1) ? Wxr : Wxh);
  const float* Wm = (g == 0) ? Whz : ((g == 1) ? Whr : Whh);
  u32* dst = (g == 0) ? Wzp : ((g == 1) ? Wrp : Whp);
  const int kp0 = blockIdx.x * 64;
  const int n0  = blockIdx.y * 64;
#pragma unroll
  for (int i = 0; i < 16; ++i) {                   // read coalesced over n
    const int idx = i * 256 + threadIdx.x;
    const int kl = idx >> 6, nl = idx & 63;
    const int k0 = (kp0 + kl) << 1;
    const float* s = (k0 < HID) ? (Wx + (size_t)k0 * HID + (n0 + nl))
                                : (Wm + (size_t)(k0 - HID) * HID + (n0 + nl));
    tile[kl][nl] = pack2(s[0], s[HID]);            // rows k0, k0+1
  }
  __syncthreads();
#pragma unroll
  for (int i = 0; i < 16; ++i) {                   // write coalesced over kp
    const int idx = i * 256 + threadIdx.x;
    const int nl = idx >> 6, kl = idx & 63;
    dst[(size_t)(n0 + nl) * 1024 + (kp0 + kl)] = tile[kl][nl];
  }
}

// Pack x (f32) -> K-pair bf16 dwords, layout preserved (rows of 512 dwords).
__global__ __launch_bounds__(256) void pack_x(const float* __restrict__ x,
                                              u32* __restrict__ Xp) {
  const size_t i = (size_t)blockIdx.x * blockDim.x + threadIdx.x;  // pair idx
  Xp[i] = pack2(x[2 * i], x[2 * i + 1]);
}

__global__ __launch_bounds__(256) void init_state(float* Hbuf, u32* Hp, u32* bar) {
  const int i = blockIdx.x * blockDim.x + threadIdx.x;
  if (i < BATCH * HID) Hbuf[i] = 0.0f;
  if (i < BATCH * KPH) Hp[i] = 0u;                 // bf16 pair (0,0)
  if (i < 2 * TSTEPS) bar[i] = 0u;
}

// ---------------------------------------------------------------------------
// Persistent recurrent kernel. Block b owns hidden columns [16b, 16b+16);
// wave w owns batch rows [16w, 16w+16). One 16x16 C-tile per gate per wave,
// K=2048 loop of v_wmma_f32_16x16x32_bf16, all operands pre-packed bf16.
// ---------------------------------------------------------------------------
__global__ __launch_bounds__(GRUTHREADS) void gru_kernel(
    const u32* __restrict__ Xp,                    // [512*64][512] packed
    const float* __restrict__ b_z, const float* __restrict__ b_r,
    const float* __restrict__ b_h,
    const u32* __restrict__ Wzp, const u32* __restrict__ Wrp,
    const u32* __restrict__ Whp,                   // [1024 n][1024 kp]
    float* Hbuf, u32* Hp, u32* RHp, u32* bar,
    float* out) {
  const int tid   = threadIdx.x;
  const int wave  = tid >> 5;
  const int lane  = tid & 31;
  const int half  = lane >> 4;
  const int l15   = lane & 15;
  const int n0    = blockIdx.x * 16;
  const int mbase = wave * 16;
  const int col   = n0 + l15;
  const int rowbase = mbase + 8 * half;            // C-tile VGPR j <-> row

  const float bzv = b_z[col];
  const float brv = b_r[col];
  const float bhv = b_h[col];
  const u32* wzcol = Wzp + (size_t)col * 1024;
  const u32* wrcol = Wrp + (size_t)col * 1024;
  const u32* whcol = Whp + (size_t)col * 1024;
  const u32* hprow  = Hp  + (size_t)(mbase + l15) * KPH;
  const u32* rhprow = RHp + (size_t)(mbase + l15) * KPH;

  float zv[8], hv[8];

  for (int t = 0; t < TSTEPS; ++t) {
    const u32* xprow = Xp + (size_t)(t * BATCH + mbase + l15) * KPH;

    // -------- sub-step 1: Z and R gates (A fragments shared) --------
    v8f az = {}; v8f ar = {};
#pragma unroll 4
    for (int kc = 0; kc < HID; kc += 32) {         // K over x_t
      v16bf a = load_a_packed(xprow, kc, half);
      az = wmma_bf16(a, load_b_packed(wzcol, kc, half), az);
      ar = wmma_bf16(a, load_b_packed(wrcol, kc, half), ar);
    }
#pragma unroll 4
    for (int kc = 0; kc < HID; kc += 32) {         // K over H
      v16bf a = load_a_packed(hprow, kc, half);
      az = wmma_bf16(a, load_b_packed(wzcol, kc + HID, half), az);
      ar = wmma_bf16(a, load_b_packed(wrcol, kc + HID, half), ar);
    }
    if (t + 1 < TSTEPS) __builtin_prefetch(xprow + BATCH * KPH, 0, 0);

#pragma unroll
    for (int j = 0; j < 8; ++j) {
      const int row = rowbase + j;
      const float z = 1.0f / (1.0f + __expf(-(az[j] + bzv)));
      const float r = 1.0f / (1.0f + __expf(-(ar[j] + brv)));
      const float h = Hbuf[(size_t)row * HID + col];
      zv[j] = z;
      hv[j] = h;
      const float rh = r * h;
      const float rh1 = __shfl_xor(rh, 1, 32);     // neighbor column's value
      if ((lane & 1) == 0)                         // even col packs (c, c+1)
        RHp[(size_t)row * KPH + (col >> 1)] = pack2(rh, rh1);
    }
    grid_barrier(bar, 2 * t);                      // RH visible grid-wide

    // -------- sub-step 2: candidate + state update --------
    v8f ah = {};
#pragma unroll 4
    for (int kc = 0; kc < HID; kc += 32) {         // K over x_t
      v16bf a = load_a_packed(xprow, kc, half);
      ah = wmma_bf16(a, load_b_packed(whcol, kc, half), ah);
    }
#pragma unroll 4
    for (int kc = 0; kc < HID; kc += 32) {         // K over R*H
      v16bf a = load_a_packed(rhprow, kc, half);
      ah = wmma_bf16(a, load_b_packed(whcol, kc + HID, half), ah);
    }
#pragma unroll
    for (int j = 0; j < 8; ++j) {
      const int row = rowbase + j;
      const float ht = tanhf(ah[j] + bhv);
      const float hn = zv[j] * hv[j] + (1.0f - zv[j]) * ht;
      Hbuf[(size_t)row * HID + col] = hn;          // in-place: own slice only
      out[(size_t)t * (BATCH * HID) + (size_t)row * HID + col] = hn;
      if (t == TSTEPS - 1)
        out[(size_t)TSTEPS * (BATCH * HID) + (size_t)row * HID + col] = hn;
      const float hn1 = __shfl_xor(hn, 1, 32);
      if ((lane & 1) == 0)                         // keep bf16 H in sync
        Hp[(size_t)row * KPH + (col >> 1)] = pack2(hn, hn1);
    }
    grid_barrier(bar, 2 * t + 1);                  // H update visible
  }
}

extern "C" void kernel_launch(void* const* d_in, const int* in_sizes, int n_in,
                              void* d_out, int out_size, void* d_ws, size_t ws_size,
                              hipStream_t stream) {
  const float* x   = (const float*)d_in[0];
  const float* Wxz = (const float*)d_in[1];
  const float* Whz = (const float*)d_in[2];
  const float* bz  = (const float*)d_in[3];
  const float* Wxr = (const float*)d_in[4];
  const float* Whr = (const float*)d_in[5];
  const float* br  = (const float*)d_in[6];
  const float* Wxh = (const float*)d_in[7];
  const float* Whh = (const float*)d_in[8];
  const float* bh  = (const float*)d_in[9];

  char* ws = (char*)d_ws;
  u32*   Wzp  = (u32*)(ws);                            // 4 MiB each
  u32*   Wrp  = (u32*)(ws + (size_t)(4 << 20));
  u32*   Whp  = (u32*)(ws + (size_t)(8 << 20));
  u32*   Xp   = (u32*)(ws + (size_t)(12 << 20));       // 64 MiB packed x
  float* Hbuf = (float*)(ws + (size_t)(76 << 20));     // 256 KiB f32 H
  u32*   Hp   = (u32*)(Hbuf + BATCH * HID);            // 128 KiB packed H
  u32*   RHp  = Hp + BATCH * KPH;                      // 128 KiB packed R*H
  u32*   bar  = RHp + BATCH * KPH;                     // 4 KiB counters

  // 1) pack [W_x;W_h] per gate, column-major K-pair bf16 (L2-resident)
  pack_weights<<<dim3(16, 16, 3), 256, 0, stream>>>(
      Wxz, Whz, Wxr, Whr, Wxh, Whh, Wzp, Wrp, Whp);
  // 2) pack x once: 16.7M bf16 pairs, streaming
  pack_x<<<(TSTEPS * BATCH * KPH) / 256, 256, 0, stream>>>(x, Xp);
  // 3) zero H0 (f32 + packed) and barrier counters (fresh -> replay safe)
  init_state<<<(BATCH * HID + 255) / 256, 256, 0, stream>>>(Hbuf, Hp, bar);
  // 4) persistent recurrent kernel
  gru_kernel<<<NBLOCKS, GRUTHREADS, 0, stream>>>(
      Xp, bz, br, bh, Wzp, Wrp, Whp, Hbuf, Hp, RHp, bar, (float*)d_out);
}
